// SparseLocalAttention_77094662964013
// MI455X (gfx1250) — matrix operand order repaired and checked by reading, not compiled, not run on hardware
//
#include <hip/hip_runtime.h>
#include <math.h>

typedef __bf16 bf16_t;
typedef bf16_t v16bf __attribute__((ext_vector_type(16)));
typedef bf16_t v8bf  __attribute__((ext_vector_type(8)));
typedef float  v8f   __attribute__((ext_vector_type(8)));
typedef float  v4f   __attribute__((ext_vector_type(4)));

#define S_LEN   2048
#define BATCH   2
#define EMB     512
#define HEADS   8
#define HDIM    64
#define WINSZ   128
#define ROWS    (S_LEN*BATCH)      // 4096
#define NQTILES (S_LEN/16)         // 128
#define KTILES  18                 // 17 band tiles + 1 padding tile (288 cols)
#define SCPAD   296                // LDS row stride (multiple of 8 halves)

static __device__ __forceinline__ v16bf cat16(v8bf lo, v8bf hi) {
  return __builtin_shufflevector(lo, hi, 0,1,2,3,4,5,6,7,8,9,10,11,12,13,14,15);
}

// A-operand chunk (16x32 bf16): lane row = lane&15,
// halves 0..15 = K[k0+ksel*16 .. k0+ksel*16+15] (contiguous, matches B)
static __device__ __forceinline__ v16bf load_a_chunk(const bf16_t* rowp, int k0, int ksel) {
  v8bf lo = *(const v8bf*)(rowp + k0 + ksel*16);
  v8bf hi = *(const v8bf*)(rowp + k0 + ksel*16 + 8);
  return cat16(lo, hi);
}

static __device__ __forceinline__ v8f wmma_bf16(v16bf a, v16bf b, v8f c) {
  return __builtin_amdgcn_wmma_f32_16x16x32_bf16(false, a, false, b, (short)0, c, false, false);
}

// ---------------------------------------------------------------- convert
__global__ void cvt_f32_bf16(const float* __restrict__ src, bf16_t* __restrict__ dst, int n) {
  int i = blockIdx.x * blockDim.x + threadIdx.x;
  int stride = gridDim.x * blockDim.x;
  for (; i < n; i += stride) dst[i] = (bf16_t)src[i];
}

// ---------------------------------------------------------------- QKV projection
// y = X @ Win^T + bias ; 16x64 output tile per wave (A reused across 4 N-tiles).
// A 64-wide column group stays inside one matrix (q/k/v) and one head (64==HDIM).
// q scaled by 1/8, stored [B,H,S,D]; k stored [B,H,S,D]; v stored transposed [B,H,D,S]
__global__ __launch_bounds__(256) void qkv_proj_kernel(
    const bf16_t* __restrict__ xq, const bf16_t* __restrict__ xk,
    const bf16_t* __restrict__ xv, const bf16_t* __restrict__ win,
    const float*  __restrict__ bias,
    bf16_t* __restrict__ qo, bf16_t* __restrict__ ko, bf16_t* __restrict__ vto)
{
  const int lane = threadIdx.x;
  const int wave = threadIdx.y;
  const int tile = blockIdx.x * 8 + wave;     // 6144 tiles (256 x 24)
  const int NCT  = (3*EMB)/64;                // 24
  const int trow = tile / NCT;
  const int tcol = tile % NCT;
  const int j0   = tcol * 64;                 // first of 64 output columns
  const int mat  = j0 / EMB;                  // 0=q 1=k 2=v (uniform per tile)
  const int h    = (j0 & (EMB-1)) >> 6;       // head (uniform per tile)
  const bf16_t* X = (mat == 0) ? xq : (mat == 1) ? xk : xv;

  const int ksel = lane >> 4;
  const int col  = lane & 15;
  const bf16_t* arowp = X   + (trow*16 + col) * EMB;      // A row = lane&15
  const bf16_t* browp = win + (j0 + col) * EMB + ksel*16; // B: contiguous along K

  v8f acc0 = {}, acc1 = {}, acc2 = {}, acc3 = {};
  for (int kc = 0; kc < EMB/32; ++kc) {
    int k0 = kc*32;
    v16bf a  = load_a_chunk(arowp, k0, ksel);
    v16bf b0 = *(const v16bf*)(browp + k0);
    v16bf b1 = *(const v16bf*)(browp + 16*EMB + k0);
    v16bf b2 = *(const v16bf*)(browp + 32*EMB + k0);
    v16bf b3 = *(const v16bf*)(browp + 48*EMB + k0);
    acc0 = wmma_bf16(a, b0, acc0);
    acc1 = wmma_bf16(a, b1, acc1);
    acc2 = wmma_bf16(a, b2, acc2);
    acc3 = wmma_bf16(a, b3, acc3);
  }

  const float bj0 = bias[j0 +  0 + col];
  const float bj1 = bias[j0 + 16 + col];
  const float bj2 = bias[j0 + 32 + col];
  const float bj3 = bias[j0 + 48 + col];
#pragma unroll
  for (int r = 0; r < 8; ++r) {
    int rowi = trow*16 + r + ksel*8;          // D rows: lanes16-31 -> M+8
    int s = rowi >> 1, b = rowi & 1;          // x-row = s*B + b
    int bh = b * HEADS + h;
    float v0 = acc0[r] + bj0, v1 = acc1[r] + bj1;
    float v2 = acc2[r] + bj2, v3 = acc3[r] + bj3;
    if (mat == 0) {
      bf16_t* p = qo + (bh*S_LEN + s)*HDIM;
      p[ 0 + col] = (bf16_t)(v0 * 0.125f);
      p[16 + col] = (bf16_t)(v1 * 0.125f);
      p[32 + col] = (bf16_t)(v2 * 0.125f);
      p[48 + col] = (bf16_t)(v3 * 0.125f);
    } else if (mat == 1) {
      bf16_t* p = ko + (bh*S_LEN + s)*HDIM;
      p[ 0 + col] = (bf16_t)v0;
      p[16 + col] = (bf16_t)v1;
      p[32 + col] = (bf16_t)v2;
      p[48 + col] = (bf16_t)v3;
    } else {
      bf16_t* p = vto + (size_t)bh*HDIM*S_LEN + s;
      p[(size_t)( 0 + col)*S_LEN] = (bf16_t)v0;
      p[(size_t)(16 + col)*S_LEN] = (bf16_t)v1;
      p[(size_t)(32 + col)*S_LEN] = (bf16_t)v2;
      p[(size_t)(48 + col)*S_LEN] = (bf16_t)v3;
    }
  }
}

// ---------------------------------------------------------------- banded attention
// One wave per (b,h, 16-row query tile). 18 key tiles (17 band + 1 pad).
__global__ __launch_bounds__(32) void attn_kernel(
    const bf16_t* __restrict__ qm, const bf16_t* __restrict__ km,
    const bf16_t* __restrict__ vt, bf16_t* __restrict__ ctx,
    float* __restrict__ attw)
{
  __shared__ __align__(16) float sc[16][SCPAD];
  __shared__ __align__(16) bf16_t pp[16][SCPAD];
  __shared__ float red[2][2][16];

  const int lane = threadIdx.x;
  const int qt   = blockIdx.x;                // 0..127
  const int bh   = blockIdx.y;                // 0..15
  const int b    = bh >> 3;
  const int h    = bh & 7;
  const int q0   = qt * 16;
  const int ksel = lane >> 4;
  const int col  = lane & 15;

  // Q tile A-operand (16x64 -> two K=32 chunks), row = lane&15
  const bf16_t* qrow = qm + (bh*S_LEN + q0 + col) * HDIM;
  v16bf qa0 = load_a_chunk(qrow, 0,  ksel);
  v16bf qa1 = load_a_chunk(qrow, 32, ksel);

  // ---- scores: S = Q K^T, masked to the band, staged in LDS
  for (int t = 0; t < KTILES; ++t) {
    int ktg = qt - 8 + t;
    bool valid = (ktg >= 0) && (ktg < NQTILES);
    v8f c = {};
    if (valid) {
      const bf16_t* kb = km + (bh*S_LEN + ktg*16 + col) * HDIM + ksel*16;
      v16bf b0 = *(const v16bf*)(kb);
      v16bf b1 = *(const v16bf*)(kb + 32);
      c = wmma_bf16(qa0, b0, c);
      c = wmma_bf16(qa1, b1, c);
    }
#pragma unroll
    for (int r = 0; r < 8; ++r) {
      int rit = r + ksel*8;
      int sq  = q0 + rit;
      int sk  = ktg*16 + col;
      bool inband = valid && (sk >= sq - WINSZ) && (sk <= sq + WINSZ);
      sc[rit][t*16 + col] = inband ? c[r] : -3.0e38f;
    }
  }
  __syncthreads();

  // ---- softmax over the 288-wide band (2 lanes per row, 144 cols each, float4)
  const int row  = lane & 15;
  const int half = lane >> 4;
  const v4f* rowv = (const v4f*)&sc[row][0];

  float lm = -3.0e38f;
  for (int c4 = half*36; c4 < half*36 + 36; ++c4) {
    v4f x = rowv[c4];
    lm = fmaxf(lm, fmaxf(fmaxf(x[0], x[1]), fmaxf(x[2], x[3])));
  }
  red[0][half][row] = lm;
  __syncthreads();
  float m = fmaxf(red[0][0][row], red[0][1][row]);

  float ls = 0.0f;
  for (int c4 = half*36; c4 < half*36 + 36; ++c4) {
    v4f x = rowv[c4];
    ls += __expf(x[0]-m) + __expf(x[1]-m) + __expf(x[2]-m) + __expf(x[3]-m);
  }
  red[1][half][row] = ls;
  __syncthreads();
  float inv_l = 1.0f / (red[1][0][row] + red[1][1][row]);

  int sq = q0 + row;
  for (int c4 = half*36; c4 < half*36 + 36; ++c4) {
    v4f x = rowv[c4];
#pragma unroll
    for (int e = 0; e < 4; ++e) {
      int c = c4*4 + e;
      float p = __expf(x[e] - m) * inv_l;     // exactly 0 where masked
      pp[row][c] = (bf16_t)p;
      int sk = (qt - 8)*16 + c;
      if (sk >= 0 && sk < S_LEN && sk >= sq - WINSZ && sk <= sq + WINSZ)
        atomicAdd(&attw[((size_t)b*S_LEN + sq)*S_LEN + sk], p * 0.125f); // mean over H=8
    }
  }
  __syncthreads();

  // ---- ctx = P @ V  (K-dim = 288 in 9 chunks of 32, V^T gives contiguous B loads)
  v8f o0 = {}, o1 = {}, o2 = {}, o3 = {};
  for (int kk = 0; kk < 9; ++kk) {
    int k0 = kk*32;
    v8bf plo = *(const v8bf*)(&pp[row][k0 + ksel*16]);
    v8bf phi = *(const v8bf*)(&pp[row][k0 + ksel*16 + 8]);
    v16bf a = cat16(plo, phi);
    int ksg = (qt - 8)*16 + k0 + ksel*16;
    int ksc = min(max(ksg, 0), S_LEN - 16);     // clamp: P==0 there, finite*0 = 0
    const bf16_t* vb = vt + (size_t)bh*HDIM*S_LEN + ksc;
    v16bf b0 = *(const v16bf*)(vb + (size_t)( 0 + col)*S_LEN);
    v16bf b1 = *(const v16bf*)(vb + (size_t)(16 + col)*S_LEN);
    v16bf b2 = *(const v16bf*)(vb + (size_t)(32 + col)*S_LEN);
    v16bf b3 = *(const v16bf*)(vb + (size_t)(48 + col)*S_LEN);
    o0 = wmma_bf16(a, b0, o0);
    o1 = wmma_bf16(a, b1, o1);
    o2 = wmma_bf16(a, b2, o2);
    o3 = wmma_bf16(a, b3, o3);
  }

#pragma unroll
  for (int r = 0; r < 8; ++r) {
    int sqr  = q0 + r + ksel*8;
    int xrow = sqr * BATCH + b;
    bf16_t* cp = ctx + (size_t)xrow*EMB + h*HDIM;
    cp[ 0 + col] = (bf16_t)o0[r];
    cp[16 + col] = (bf16_t)o1[r];
    cp[32 + col] = (bf16_t)o2[r];
    cp[48 + col] = (bf16_t)o3[r];
  }
}

// ---------------------------------------------------------------- output projection
// 16x64 output tile per wave, A reused across 4 N-tiles.
__global__ __launch_bounds__(256) void out_proj_kernel(
    const bf16_t* __restrict__ ctx, const bf16_t* __restrict__ wout,
    const float* __restrict__ bias, float* __restrict__ out)
{
  const int lane = threadIdx.x;
  const int wave = threadIdx.y;
  const int tile = blockIdx.x * 8 + wave;     // 2048 tiles (256 x 8)
  const int NCT  = EMB/64;                    // 8
  const int trow = tile / NCT;
  const int tcol = tile % NCT;
  const int j0   = tcol * 64;
  const int ksel = lane >> 4;
  const int col  = lane & 15;

  const bf16_t* arowp = ctx  + (trow*16 + col) * EMB;
  const bf16_t* browp = wout + (j0 + col) * EMB + ksel*16;

  v8f acc0 = {}, acc1 = {}, acc2 = {}, acc3 = {};
  for (int kc = 0; kc < EMB/32; ++kc) {
    int k0 = kc*32;
    v16bf a  = load_a_chunk(arowp, k0, ksel);
    v16bf b0 = *(const v16bf*)(browp + k0);
    v16bf b1 = *(const v16bf*)(browp + 16*EMB + k0);
    v16bf b2 = *(const v16bf*)(browp + 32*EMB + k0);
    v16bf b3 = *(const v16bf*)(browp + 48*EMB + k0);
    acc0 = wmma_bf16(a, b0, acc0);
    acc1 = wmma_bf16(a, b1, acc1);
    acc2 = wmma_bf16(a, b2, acc2);
    acc3 = wmma_bf16(a, b3, acc3);
  }

  const float bj0 = bias[j0 +  0 + col];
  const float bj1 = bias[j0 + 16 + col];
  const float bj2 = bias[j0 + 32 + col];
  const float bj3 = bias[j0 + 48 + col];
#pragma unroll
  for (int r = 0; r < 8; ++r) {
    int rowi = trow*16 + r + ksel*8;
    float* op = out + (size_t)rowi*EMB + j0;
    op[ 0 + col] = acc0[r] + bj0;
    op[16 + col] = acc1[r] + bj1;
    op[32 + col] = acc2[r] + bj2;
    op[48 + col] = acc3[r] + bj3;
  }
}

// ---------------------------------------------------------------- launch
extern "C" void kernel_launch(void* const* d_in, const int* in_sizes, int n_in,
                              void* d_out, int out_size, void* d_ws, size_t ws_size,
                              hipStream_t stream) {
  const float* query  = (const float*)d_in[0];   // [S,B,E]
  const float* key_t  = (const float*)d_in[1];
  const float* value  = (const float*)d_in[2];
  const float* w_in   = (const float*)d_in[3];   // [3E,E]
  const float* b_in   = (const float*)d_in[4];   // [3E]
  const float* w_out  = (const float*)d_in[5];   // [E,E]
  const float* b_out  = (const float*)d_in[6];   // [E]

  // workspace layout (bytes)
  char* ws = (char*)d_ws;
  const size_t XN  = (size_t)ROWS * EMB;          // 2,097,152 elements
  bf16_t* xq  = (bf16_t*)(ws);                    //  4 MB
  bf16_t* xk  = (bf16_t*)(ws +  4u*1024*1024);
  bf16_t* xv  = (bf16_t*)(ws +  8u*1024*1024);
  bf16_t* wi  = (bf16_t*)(ws + 12u*1024*1024);    // 1536*512 bf16 = 1.5 MB
  bf16_t* wo  = (bf16_t*)(ws + 14u*1024*1024);    //  512*512 bf16 = 0.5 MB
  bf16_t* qbh = (bf16_t*)(ws + 15u*1024*1024);    // [B,H,S,D]  4 MB
  bf16_t* kbh = (bf16_t*)(ws + 19u*1024*1024);    // [B,H,S,D]  4 MB
  bf16_t* vtb = (bf16_t*)(ws + 23u*1024*1024);    // [B,H,D,S]  4 MB
  bf16_t* ctx = (bf16_t*)(ws + 27u*1024*1024);    // [4096,512] 4 MB

  float* out_f  = (float*)d_out;                  // [S,B,E]
  float* attn_w = out_f + XN;                     // [B,S,S]

  // 1) one-time f32 -> bf16 conversions
  cvt_f32_bf16<<<2048, 256, 0, stream>>>(query, xq, (int)XN);
  cvt_f32_bf16<<<2048, 256, 0, stream>>>(key_t, xk, (int)XN);
  cvt_f32_bf16<<<2048, 256, 0, stream>>>(value, xv, (int)XN);
  cvt_f32_bf16<<<1024, 256, 0, stream>>>(w_in,  wi, 3*EMB*EMB);
  cvt_f32_bf16<<< 512, 256, 0, stream>>>(w_out, wo, EMB*EMB);

  // 2) zero attn_weights (band atomics land on top)
  hipMemsetAsync(attn_w, 0, (size_t)BATCH * S_LEN * S_LEN * sizeof(float), stream);

  // 3) fused QKV projection: 6144 16x64 wave-tiles, 8 waves/block
  qkv_proj_kernel<<<768, dim3(32, 8), 0, stream>>>(xq, xk, xv, wi, b_in, qbh, kbh, vtb);

  // 4) banded attention: one wave per (qtile, b*h)
  attn_kernel<<<dim3(NQTILES, BATCH*HEADS), 32, 0, stream>>>(qbh, kbh, vtb, ctx, attn_w);

  // 5) output projection: 2048 16x64 wave-tiles, 8 waves/block
  out_proj_kernel<<<256, dim3(32, 8), 0, stream>>>(ctx, wo, b_out, out_f);
}
